// AttentionFunctionNode_4466765988074
// MI455X (gfx1250) — matrix operand order repaired and checked by reading, not compiled, not run on hardware
//
#include <hip/hip_runtime.h>
#include <hip/hip_bf16.h>

// ---------------- problem constants ----------------
constexpr int D   = 768;     // feature dim
constexpr int NP  = 2048;    // patches per batch
constexpr int BB  = 32;      // batches
constexpr int BM  = 32;      // query rows per workgroup (2 blocks of 16)
constexpr int BN  = 64;      // key rows per (double-buffered) tile
constexpr int NWAVE = 8;     // 256 threads / wave32

constexpr int VSTR = BN + 8;   // 72 : Vt row stride (bf16 elems), padded
constexpr int QSTR = D + 16;   // 784: Q row stride (bf16 elems)
constexpr int PSTR = BM + 8;   // 40 : Pt row stride (32 queries + pad)

constexpr float SCALE = 0.03608439182435161f;  // 768^-0.5

constexpr int VBYTES = 2 * D * VSTR * 2;    // 221184 (two buffers)
constexpr int QBYTES = BM * QSTR * 2;       // 50176
constexpr int PBYTES = BN * PSTR * 2;       // 5120
constexpr int RBYTES = (3 * 32 + 2 * 8 * 32) * 4;  // 2432
constexpr int SMEM   = VBYTES + QBYTES + PBYTES + RBYTES;  // 278912

typedef __bf16 bf16;
typedef bf16  v16bf __attribute__((ext_vector_type(16)));
typedef bf16  v8bf  __attribute__((ext_vector_type(8)));
typedef float v8f   __attribute__((ext_vector_type(8)));

union BFV { v16bf v; v8bf h[2]; };

static __device__ __forceinline__ unsigned pack2(float a, float b) {
    unsigned short ua = __builtin_bit_cast(unsigned short, (bf16)a);
    unsigned short ub = __builtin_bit_cast(unsigned short, (bf16)b);
    return (unsigned)ua | ((unsigned)ub << 16);
}

// convert + transpose one 64-key tile: x[kb..kb+63][0..767] fp32 -> Vt[feat][key] bf16
// 8 keys x 4 features per task -> four ds_store_b128 per 32 elements
static __device__ __forceinline__ void fill_tile(bf16* __restrict__ Vt,
                                                 const float* __restrict__ xk,
                                                 int tid) {
    const float4* xk4 = (const float4*)xk;
    for (int task = tid; task < (BN / 8) * (D / 4); task += 256) {
        int grp = task / (D / 4);      // key octet
        int q4  = task % (D / 4);      // feature quad
        int k0  = grp * 8;
        float4 v0 = xk4[(size_t)(k0 + 0) * (D / 4) + q4];
        float4 v1 = xk4[(size_t)(k0 + 1) * (D / 4) + q4];
        float4 v2 = xk4[(size_t)(k0 + 2) * (D / 4) + q4];
        float4 v3 = xk4[(size_t)(k0 + 3) * (D / 4) + q4];
        float4 v4 = xk4[(size_t)(k0 + 4) * (D / 4) + q4];
        float4 v5 = xk4[(size_t)(k0 + 5) * (D / 4) + q4];
        float4 v6 = xk4[(size_t)(k0 + 6) * (D / 4) + q4];
        float4 v7 = xk4[(size_t)(k0 + 7) * (D / 4) + q4];
        int f = q4 * 4;
        uint4 u;
        u.x = pack2(v0.x, v1.x); u.y = pack2(v2.x, v3.x);
        u.z = pack2(v4.x, v5.x); u.w = pack2(v6.x, v7.x);
        *(uint4*)&Vt[(f + 0) * VSTR + k0] = u;
        u.x = pack2(v0.y, v1.y); u.y = pack2(v2.y, v3.y);
        u.z = pack2(v4.y, v5.y); u.w = pack2(v6.y, v7.y);
        *(uint4*)&Vt[(f + 1) * VSTR + k0] = u;
        u.x = pack2(v0.z, v1.z); u.y = pack2(v2.z, v3.z);
        u.z = pack2(v4.z, v5.z); u.w = pack2(v6.z, v7.z);
        *(uint4*)&Vt[(f + 2) * VSTR + k0] = u;
        u.x = pack2(v0.w, v1.w); u.y = pack2(v2.w, v3.w);
        u.z = pack2(v4.w, v5.w); u.w = pack2(v6.w, v7.w);
        *(uint4*)&Vt[(f + 3) * VSTR + k0] = u;
    }
}

__global__ __launch_bounds__(256, 1)
void attn_flash_wmma(const float* __restrict__ x, float* __restrict__ out) {
    extern __shared__ char smem[];
    bf16*  Vt0 = (bf16*)smem;                               // [D][VSTR] buffer 0
    bf16*  Vt1 = (bf16*)(smem + D * VSTR * 2);              // [D][VSTR] buffer 1
    bf16*  Qr  = (bf16*)(smem + VBYTES);                    // [BM][QSTR] row major
    bf16*  Pt  = (bf16*)(smem + VBYTES + QBYTES);           // [BN][PSTR] (P transposed)
    float* red = (float*)(smem + VBYTES + QBYTES + PBYTES);
    float* rowmax  = red;            // [32]
    float* rowsum  = red + 32;       // [32]
    float* corrf   = red + 64;       // [32]
    float* wavemax = red + 96;       // [8][32]
    float* wavesum = red + 96 + 256; // [8][32]

    const int tid   = threadIdx.x;
    const int wave  = tid >> 5;
    const int lane  = tid & 31;
    const int lhalf = (lane >> 4);   // 0: lanes 0-15, 1: lanes 16-31
    const int l16   = lane & 15;
    const int acol  = lhalf * 8;     // A-operand K sub-offset for this half

    const int r  = wave >> 2;        // query block (0/1) this wave scores
    const int kt = wave & 3;         // key tile (16 cols) this wave scores
    const int rb = r * 16;

    const int wg   = blockIdx.x;
    const int bidx = wg / (NP / BM);
    const int qb   = (wg % (NP / BM)) * BM;
    const float* xb = x + (size_t)bidx * NP * D;

    // ---- load + convert Q tile (fp32 -> bf16, row major) ----
    {
        const float4* xq4 = (const float4*)(xb + (size_t)qb * D);
        for (int task = tid; task < BM * (D / 4); task += 256) {
            int row = task / (D / 4);
            int q4  = task % (D / 4);
            float4 v = xq4[row * (D / 4) + q4];
            unsigned* dst = (unsigned*)&Qr[row * QSTR + q4 * 4];
            dst[0] = pack2(v.x, v.y);
            dst[1] = pack2(v.z, v.w);
        }
    }
    if (tid < 32) { rowmax[tid] = -1e30f; rowsum[tid] = 0.0f; }

    // prologue: fill buffer 0 with key tile 0
    fill_tile(Vt0, xb, tid);
    __syncthreads();

    v8f acc[6][2] = {};   // O^T accumulators: 6 feat tiles x 2 query blocks

    for (int it = 0; it < NP / BN; ++it) {
        const int kb = it * BN;
        bf16* cur = (it & 1) ? Vt1 : Vt0;
        bf16* nxt = (it & 1) ? Vt0 : Vt1;

        // ---- pipeline: start fetching + converting the NEXT tile now ----
        if (it + 1 < NP / BN) {
            fill_tile(nxt, xb + (size_t)(kb + BN) * D, tid);
            if (kb + 2 * BN < NP) {   // prefetch tile after next into L2
                const char* pf = (const char*)(xb + (size_t)(kb + 2 * BN) * D);
                #pragma unroll
                for (int i = 0; i < 3; ++i)
                    __builtin_prefetch(pf + ((size_t)tid + 256 * i) * 256, 0, 1);
            }
        }

        // ---- S tile: queries [qb+rb, +16) x keys [kb+16*kt, +16) ----
        v8f c = {};
        {
            const bf16* qrow = Qr + (rb + l16) * QSTR;
            #pragma unroll 4
            for (int kc = 0; kc < D / 32; ++kc) {
                BFV A, Bm;
                A.h[0] = *(const v8bf*)(qrow + kc * 32 + acol);
                A.h[1] = *(const v8bf*)(qrow + kc * 32 + 16 + acol);
                const bf16* vcol = cur + (kc * 32 + acol) * VSTR + kt * 16 + l16;
                #pragma unroll
                for (int s = 0; s < 8; ++s) {
                    Bm.v[s]     = vcol[s * VSTR];
                    Bm.v[8 + s] = vcol[(16 + s) * VSTR];
                }
                c = __builtin_amdgcn_wmma_f32_16x16x32_bf16(
                        false, A.v, false, Bm.v, (short)0, c, false, false);
            }
        }

        // ---- per-row (query) max over this wave's 16 keys ----
        float rm[8];
        #pragma unroll
        for (int g = 0; g < 8; ++g) {
            float s = c[g] * SCALE;
            c[g] = s;
            float m = s;
            #pragma unroll
            for (int off = 8; off >= 1; off >>= 1)
                m = fmaxf(m, __shfl_xor(m, off, 32));
            rm[g] = m;
        }
        if (l16 == 0) {
            #pragma unroll
            for (int g = 0; g < 8; ++g)
                wavemax[wave * 32 + rb + lhalf * 8 + g] = rm[g];
        }
        __syncthreads();

        // ---- combine block max across the 4 waves covering each row ----
        if (tid < 32) {
            int rr = tid >> 4;                  // query block of this row
            float mx = rowmax[tid];
            #pragma unroll
            for (int w = 0; w < 4; ++w)
                mx = fmaxf(mx, wavemax[(rr * 4 + w) * 32 + tid]);
            corrf[tid]  = __expf(rowmax[tid] - mx);
            rowmax[tid] = mx;
        }
        __syncthreads();

        // ---- exp, partial sums, P^T store, rescale O accumulators ----
        float nmv[8];
        #pragma unroll
        for (int g = 0; g < 8; ++g) nmv[g] = rowmax[rb + lhalf * 8 + g];
        float cf0 = corrf[l16];
        float cf1 = corrf[16 + l16];
        // skip the 96-FLOP rescale when no row max changed (corrf == 1 exactly);
        // wave-uniform ballot -> scalar branch
        if (__any((cf0 < 1.0f) || (cf1 < 1.0f))) {
            #pragma unroll
            for (int t = 0; t < 6; ++t) {
                #pragma unroll
                for (int g = 0; g < 8; ++g) {
                    acc[t][0][g] *= cf0;
                    acc[t][1][g] *= cf1;
                }
            }
        }

        float psum[8];
        #pragma unroll
        for (int g = 0; g < 8; ++g) {
            float p = __expf(c[g] - nmv[g]);
            c[g] = p;
            float s = p;
            #pragma unroll
            for (int off = 8; off >= 1; off >>= 1)
                s += __shfl_xor(s, off, 32);
            psum[g] = s;
            Pt[(kt * 16 + l16) * PSTR + rb + lhalf * 8 + g] = (bf16)p;
        }
        if (l16 == 0) {
            #pragma unroll
            for (int g = 0; g < 8; ++g)
                wavesum[wave * 32 + rb + lhalf * 8 + g] = psum[g];
        }
        __syncthreads();

        if (tid < 32) {
            int rr = tid >> 4;
            float s = rowsum[tid] * corrf[tid];
            #pragma unroll
            for (int w = 0; w < 4; ++w)
                s += wavesum[(rr * 4 + w) * 32 + tid];
            rowsum[tid] = s;
        }

        // ---- O^T += V^T x P^T : wave owns feature slice [96*wave, +96) ----
        #pragma unroll
        for (int t = 0; t < 6; ++t) {
            const bf16* arow = cur + (wave * 96 + t * 16 + l16) * VSTR;
            #pragma unroll
            for (int kc = 0; kc < BN / 32; ++kc) {
                BFV A;
                A.h[0] = *(const v8bf*)(arow + kc * 32 + acol);
                A.h[1] = *(const v8bf*)(arow + kc * 32 + 16 + acol);
                #pragma unroll
                for (int r2 = 0; r2 < 2; ++r2) {
                    BFV Bm;
                    const bf16* pcol = Pt + (kc * 32 + acol) * PSTR + r2 * 16 + l16;
                    #pragma unroll
                    for (int s = 0; s < 8; ++s) {
                        Bm.v[s]     = pcol[s * PSTR];
                        Bm.v[8 + s] = pcol[(16 + s) * PSTR];
                    }
                    acc[t][r2] = __builtin_amdgcn_wmma_f32_16x16x32_bf16(
                                     false, A.v, false, Bm.v, (short)0,
                                     acc[t][r2], false, false);
                }
            }
        }
        __syncthreads();   // gates buffer swap + Pt reuse
    }

    // ---- finalize: divide by softmax sum, write O ----
    #pragma unroll
    for (int r2 = 0; r2 < 2; ++r2) {
        float inv = 1.0f / rowsum[r2 * 16 + l16];
        float* orow = out + ((size_t)bidx * NP + qb + r2 * 16 + l16) * D;
        #pragma unroll
        for (int t = 0; t < 6; ++t) {
            int fbase = wave * 96 + t * 16 + lhalf * 8;
            float4 v0 = make_float4(acc[t][r2][0] * inv, acc[t][r2][1] * inv,
                                    acc[t][r2][2] * inv, acc[t][r2][3] * inv);
            float4 v1 = make_float4(acc[t][r2][4] * inv, acc[t][r2][5] * inv,
                                    acc[t][r2][6] * inv, acc[t][r2][7] * inv);
            *(float4*)(orow + fbase)     = v0;
            *(float4*)(orow + fbase + 4) = v1;
        }
    }
}

extern "C" void kernel_launch(void* const* d_in, const int* in_sizes, int n_in,
                              void* d_out, int out_size, void* d_ws, size_t ws_size,
                              hipStream_t stream) {
    (void)in_sizes; (void)n_in; (void)d_ws; (void)ws_size; (void)out_size;
    const float* x = (const float*)d_in[0];
    float* out = (float*)d_out;

    // allow >64KB dynamic LDS (CDNA5 WGP has 320KB)
    hipFuncSetAttribute((const void*)attn_flash_wmma,
                        hipFuncAttributeMaxDynamicSharedMemorySize, SMEM);

    dim3 grid(BB * (NP / BM));   // 32 * 64 = 2048 workgroups
    dim3 block(256);             // 8 wave32
    attn_flash_wmma<<<grid, block, SMEM, stream>>>(x, out);
}
